// DigitalRockINR_10273561772149
// MI455X (gfx1250) — matrix operand order repaired and checked by reading, not compiled, not run on hardware
//
#include <hip/hip_runtime.h>
#include <hip/hip_bf16.h>

// ---- problem constants (match reference) ----
#define NLVL 16
#define LOG2_T 19
#define TBL (1u << LOG2_T)
#define HID 64

// resolutions = ceil(16 * 2^(i/3))  (growth factor b = 32^(1/15) = 2^(1/3))
constexpr int RES[NLVL] = {16, 21, 26, 32, 41, 51, 64, 81, 102, 128,
                           162, 204, 256, 323, 407, 512};

typedef __attribute__((ext_vector_type(16))) _Float16 v16h;
typedef __attribute__((ext_vector_type(8)))  _Float16 v8h;
typedef __attribute__((ext_vector_type(8)))  float    v8f;

// A-fragment: two 16-byte pieces (K kb..kb+7 in VGPR0-3, K kb+16..kb+23 in VGPR4-7)
__device__ inline v16h load_a_frag(const _Float16* p) {
  v8h lo = *(const v8h*)p;
  v8h hi = *(const v8h*)(p + 16);
  v16h r;
#pragma unroll
  for (int i = 0; i < 8; ++i) { r[i] = lo[i]; r[i + 8] = hi[i]; }
  return r;
}

// Hash-encode 8 levels for one point; writes 16 f16 features into actRow[2*lvl..]
template <int LVL0>
__device__ inline void encode8(float x, float y, float z,
                               const float* __restrict__ tables,
                               _Float16* actRow) {
#pragma unroll
  for (int li = 0; li < 8; ++li) {
    const int lvl = LVL0 + li;
    const float res = (float)RES[lvl];
    float sx = x * res, sy = y * res, sz = z * res;
    float fx = floorf(sx), fy = floorf(sy), fz = floorf(sz);
    unsigned bx = (unsigned)fx, by = (unsigned)fy, bz = (unsigned)fz;
    float tx = sx - fx, ty = sy - fy, tz = sz - fz;
    const float2* tb = (const float2*)tables + (size_t)lvl * TBL;
    float f0 = 0.f, f1 = 0.f;
#pragma unroll
    for (int c = 0; c < 8; ++c) {
      unsigned ox = (c >> 2) & 1u, oy = (c >> 1) & 1u, oz = c & 1u;
      unsigned h = (bx + ox) ^ ((by + oy) * 2654435761u) ^ ((bz + oz) * 805459861u);
      unsigned idx = h & (TBL - 1u);
      float w = (ox ? tx : 1.f - tx) * (oy ? ty : 1.f - ty) * (oz ? tz : 1.f - tz);
      float2 g = tb[idx];
      f0 = fmaf(w, g.x, f0);
      f1 = fmaf(w, g.y, f1);
    }
    actRow[2 * lvl]     = (_Float16)f0;
    actRow[2 * lvl + 1] = (_Float16)f1;
  }
}

__global__ __launch_bounds__(256)
void inr_fused_kernel(const float* __restrict__ coords,
                      const float* __restrict__ tables,
                      const float* __restrict__ W0, const float* __restrict__ b0,
                      const float* __restrict__ W1, const float* __restrict__ b1,
                      const float* __restrict__ W2, const float* __restrict__ b2,
                      const float* __restrict__ W3, const float* __restrict__ b3,
                      float* __restrict__ out, int N) {
  // ---- LDS staging ----
  __shared__ __align__(16) _Float16 sW0t[HID * 32];  // [n][k] transposed, f16
  __shared__ __align__(16) _Float16 sW1t[HID * HID];
  __shared__ __align__(16) _Float16 sW2t[HID * HID];
  __shared__ float sW3[HID];
  __shared__ float sB0[HID], sB1[HID], sB2[HID];
  __shared__ float sB3;
  __shared__ __align__(16) _Float16 sAct[8 * 16 * HID];  // per-wave 16x64 activations

  const int tid  = threadIdx.x;
  const int lane = tid & 31;
  const int wave = tid >> 5;

  const int ptBase = blockIdx.x * 128 + wave * 16;  // 8 waves x 16 points

  // Prefetch this block's coordinate range into cache (global_prefetch_b8)
  __builtin_prefetch(&coords[(size_t)(blockIdx.x * 128) * 3 + tid * 3], 0, 3);

  // ---- cooperative weight load: f32 -> f16, transposed to [n][k] ----
  for (int i = tid; i < 32 * HID; i += 256) {
    int k = i >> 6, n = i & 63;
    sW0t[n * 32 + k] = (_Float16)W0[i];
  }
  for (int i = tid; i < HID * HID; i += 256) {
    int k = i >> 6, n = i & 63;
    sW1t[n * HID + k] = (_Float16)W1[i];
  }
  for (int i = tid; i < HID * HID; i += 256) {
    int k = i >> 6, n = i & 63;
    sW2t[n * HID + k] = (_Float16)W2[i];
  }
  if (tid < HID) {
    sW3[tid] = W3[tid];
    sB0[tid] = b0[tid]; sB1[tid] = b1[tid]; sB2[tid] = b2[tid];
  }
  if (tid == 0) sB3 = b3[0];

  // ---- hash encoding: 2 lanes per point (levels 0-7 / 8-15) ----
  const int m = lane & 15;                 // point within wave tile
  int point  = ptBase + m;
  int pc     = point < N ? point : N - 1;  // clamp so EXEC stays full for WMMA
  float x = coords[pc * 3 + 0];
  float y = coords[pc * 3 + 1];
  float z = coords[pc * 3 + 2];
  x = fminf(fmaxf(x, 0.f), 1.f - 1e-6f);
  y = fminf(fmaxf(y, 0.f), 1.f - 1e-6f);
  z = fminf(fmaxf(z, 0.f), 1.f - 1e-6f);

  _Float16* act = sAct + wave * 16 * HID;  // this wave's 16x64 tile
  if (lane < 16) encode8<0>(x, y, z, tables, act + m * HID);
  else           encode8<8>(x, y, z, tables, act + m * HID);
  __syncthreads();

  // ---- WMMA MLP ----
  const int kbA  = (lane >= 16) ? 8 : 0;    // A-fragment K base (per ISA layout)
  const int k0B  = (lane >= 16) ? 8 : 0;    // B-fragment K base (per ISA layout)
  const int ncol = lane & 15;               // C/D column
  const int mRow = (lane >= 16) ? 8 : 0;    // C/D row base

  // Layer 0: [16x32] @ [32x64] -> 4 output tiles
  {
    v16h a = load_a_frag(act + m * HID + kbA);  // cols 0..31 valid
    v8f  h[4];
#pragma unroll
    for (int t = 0; t < 4; ++t) {
      v16h bf = load_a_frag(sW0t + (ncol + 16 * t) * 32 + k0B);
      v8f cz = {};
      h[t] = __builtin_amdgcn_wmma_f32_16x16x32_f16(false, a, false, bf,
                                                    (short)0, cz, false, false);
    }
#pragma unroll
    for (int t = 0; t < 4; ++t) {
      float bias = sB0[ncol + 16 * t];
#pragma unroll
      for (int r = 0; r < 8; ++r) {
        float v = fmaxf(h[t][r] + bias, 0.f);
        act[(r + mRow) * HID + (ncol + 16 * t)] = (_Float16)v;
      }
    }
  }
  __syncthreads();

  // Layers 1 & 2: [16x64] @ [64x64], K split into two 32-chunks
#pragma unroll
  for (int layer = 0; layer < 2; ++layer) {
    const _Float16* Wt = (layer == 0) ? sW1t : sW2t;
    const float*    Bv = (layer == 0) ? sB1  : sB2;
    v16h a0 = load_a_frag(act + m * HID + kbA);        // K 0..31
    v16h a1 = load_a_frag(act + m * HID + 32 + kbA);   // K 32..63
    v8f h[4];
#pragma unroll
    for (int t = 0; t < 4; ++t) {
      v16h bf0 = load_a_frag(Wt + (ncol + 16 * t) * HID + k0B);
      v16h bf1 = load_a_frag(Wt + (ncol + 16 * t) * HID + 32 + k0B);
      v8f cz = {};
      v8f acc = __builtin_amdgcn_wmma_f32_16x16x32_f16(false, a0, false, bf0,
                                                       (short)0, cz, false, false);
      acc = __builtin_amdgcn_wmma_f32_16x16x32_f16(false, a1, false, bf1,
                                                   (short)0, acc, false, false);
      h[t] = acc;
    }
#pragma unroll
    for (int t = 0; t < 4; ++t) {
      float bias = Bv[ncol + 16 * t];
#pragma unroll
      for (int r = 0; r < 8; ++r) {
        float v = fmaxf(h[t][r] + bias, 0.f);
        act[(r + mRow) * HID + (ncol + 16 * t)] = (_Float16)v;
      }
    }
    __syncthreads();
  }

  // Layer 3: [16x64] @ [64x1] + sigmoid — per-lane dot (lanes 0..15 = points)
  if (lane < 16) {
    const _Float16* hrow = act + lane * HID;
    float sum = 0.f;
#pragma unroll
    for (int j = 0; j < HID; ++j) sum = fmaf((float)hrow[j], sW3[j], sum);
    float xv = sum + sB3;
    float yv = 1.0f / (1.0f + __expf(-xv));
    int p = ptBase + lane;
    if (p < N) out[p] = yv;
  }
}

extern "C" void kernel_launch(void* const* d_in, const int* in_sizes, int n_in,
                              void* d_out, int out_size, void* d_ws, size_t ws_size,
                              hipStream_t stream) {
  const float* coords = (const float*)d_in[0];
  const float* tables = (const float*)d_in[1];
  const float* W0 = (const float*)d_in[2];
  const float* b0 = (const float*)d_in[3];
  const float* W1 = (const float*)d_in[4];
  const float* b1 = (const float*)d_in[5];
  const float* W2 = (const float*)d_in[6];
  const float* b2 = (const float*)d_in[7];
  const float* W3 = (const float*)d_in[8];
  const float* b3 = (const float*)d_in[9];
  float* out = (float*)d_out;

  const int N = in_sizes[0] / 3;            // 2,000,000
  const int blocks = (N + 127) / 128;       // 128 points per 256-thread block

  inr_fused_kernel<<<blocks, 256, 0, stream>>>(coords, tables, W0, b0, W1, b1,
                                               W2, b2, W3, b3, out, N);
}
